// _CorrectionLayer_3590592659682
// MI455X (gfx1250) — compile-verified, matrix operand order checked by reading, not run on hardware
//
#include <hip/hip_runtime.h>

// Dimensions (fixed by the reference)
#define Cc   64
#define Hh   512
#define Ww   512
#define Bb   4
#define MT   (Bb * Hh)          // 2048 rows (B*H)
#define GATES 192               // 3*C
#define KTOT  256               // 192 (x @ w_ih^T) + 64 (h @ w_hh^T)
#define ROWS_PER_WG 128
#define NUM_WG (MT / ROWS_PER_WG)   // 16
#define LDS_ROW 72              // 64 bf16 + 8 pad (144 B, 16B-aligned chunks)

typedef __attribute__((ext_vector_type(16))) __bf16 v16bf;
typedef __attribute__((ext_vector_type(8)))  __bf16 v8bf;
typedef __attribute__((ext_vector_type(8)))  float  v8f;

__device__ __forceinline__ unsigned short f2bf(float f) {
  unsigned u = __float_as_uint(f);
  u += 0x7FFFu + ((u >> 16) & 1u);      // round-to-nearest-even
  return (unsigned short)(u >> 16);
}
__device__ __forceinline__ float bf2f(unsigned short s) {
  return __uint_as_float(((unsigned)s) << 16);
}

// ---------------------------------------------------------------------------
// Prep: Wcomb[n][k] bf16, k<192 -> w_ih[n][k], k>=192 -> w_hh[n][k-192]
// ---------------------------------------------------------------------------
__global__ __launch_bounds__(256) void gru_prep_weights(
    const float* __restrict__ wih, const float* __restrict__ whh,
    unsigned short* __restrict__ Wc) {
  int i = blockIdx.x * 256 + threadIdx.x;
  if (i >= GATES * KTOT) return;
  int n = i >> 8, k = i & 255;
  float v = (k < GATES) ? wih[n * GATES + k] : whh[n * Cc + (k - GATES)];
  Wc[i] = f2bf(v);
}

// ---------------------------------------------------------------------------
// Transpose features [B,C,H,W] f32 -> xT [W][B*H][C] bf16 (LDS-tiled),
// and copy column 0 (f32 passthrough) into d_out.
// grid = (W/32, B*H), block = 256
// ---------------------------------------------------------------------------
__global__ __launch_bounds__(256) void gru_transpose(
    const float* __restrict__ feat, unsigned short* __restrict__ xT,
    float* __restrict__ out) {
  __shared__ float tile[Cc][33];
  const int m  = blockIdx.y;              // b*H + h
  const int d0 = blockIdx.x * 32;
  const int b  = m >> 9, hh = m & (Hh - 1);
  const int t  = threadIdx.x;
  const int dx = t & 31, cy = t >> 5;
  #pragma unroll
  for (int c = cy; c < Cc; c += 8) {
    size_t base = (((size_t)(b * Cc + c) * Hh) + hh) * Ww;
    float v = feat[base + d0 + dx];
    tile[c][dx] = v;
    if (d0 + dx == 0) out[base] = v;      // column 0 passthrough (same flat index)
  }
  __syncthreads();
  const int j = t & 31;                   // dword within row: channels 2j, 2j+1
  unsigned* xTw = (unsigned*)xT;
  #pragma unroll
  for (int it = 0; it < 4; ++it) {
    int dr = (t >> 5) + it * 8;           // 0..31
    unsigned lo = f2bf(tile[2 * j][dr]);
    unsigned hi = f2bf(tile[2 * j + 1][dr]);
    size_t eo = (((size_t)(d0 + dr) * MT + m) * Cc + 2 * j) >> 1;
    xTw[eo] = lo | (hi << 16);
  }
}

// ---------------------------------------------------------------------------
// One scan step (column d). grid = 16 WGs x 256 threads (8 wave32 waves).
// Wave w computes rows [rowBase + 16w, +16) x all 192 gates.
//   A (16x32 bf16) from LDS-staged prev column (with +/-1 vertical halo) and
//     current original column; B (32x16 bf16) from L2-resident Wcomb.
//   96 v_wmma_f32_16x16x32_bf16 per wave.  n-gate keeps i_n / h_n split.
// Writes f32 to out[B,C,H,W] and updated bf16 state in place into xT[d].
// ---------------------------------------------------------------------------
__global__ __launch_bounds__(256) void gru_scan_step(
    const unsigned short* __restrict__ Wc,   // [192][256] bf16
    unsigned short* __restrict__ xT,         // [512][2048][64] bf16
    const float* __restrict__ bih, const float* __restrict__ bhh,
    float* __restrict__ out, int d) {
  __shared__ unsigned short sPrev[(ROWS_PER_WG + 2) * LDS_ROW];
  __shared__ unsigned short sH[ROWS_PER_WG * LDS_ROW];

  const int t = threadIdx.x;
  const int rowBase = blockIdx.x * ROWS_PER_WG;     // multiple of 128 -> one image
  const unsigned short* prevCol = xT + (size_t)(d - 1) * (MT * Cc);
  unsigned short* curCol = xT + (size_t)d * (MT * Cc);

  // Stage prev column rows [rowBase-1, rowBase+128] (zero outside image in H)
  for (int i = t; i < (ROWS_PER_WG + 2) * 8; i += 256) {
    int r = i >> 3, kc = (i & 7) * 8;
    int nh = (rowBase & (Hh - 1)) - 1 + r;          // neighbor h within image
    uint4 v = make_uint4(0u, 0u, 0u, 0u);
    if ((unsigned)nh < (unsigned)Hh)
      v = *(const uint4*)(prevCol + (size_t)(rowBase - 1 + r) * Cc + kc);
    *(uint4*)(sPrev + r * LDS_ROW + kc) = v;
  }
  // Stage current (original) column rows
  for (int i = t; i < ROWS_PER_WG * 8; i += 256) {
    int r = i >> 3, kc = (i & 7) * 8;
    *(uint4*)(sH + r * LDS_ROW + kc) =
        *(const uint4*)(curCol + (size_t)(rowBase + r) * Cc + kc);
  }
  __syncthreads();

  const int wv   = t >> 5;
  const int lane = t & 31;
  const int ln   = lane & 15;     // N (or M) position within tile
  const int half = lane >> 4;
  const int mloc = wv * 16;       // this wave's local row base
  const int rowA = mloc + ln;     // A-operand row for this lane
  const int kb   = half * 8;      // per-lane K sub-offset (ISA A/B layout)

  v8f accRZ[8], accI[4], accH[4];
  #pragma unroll
  for (int jj = 0; jj < 8; ++jj) {
    float bsum = bih[jj * 16 + ln] + bhh[jj * 16 + ln];
    #pragma unroll
    for (int e = 0; e < 8; ++e) accRZ[jj][e] = bsum;
  }
  #pragma unroll
  for (int jj = 0; jj < 4; ++jj) {
    float bi = bih[128 + jj * 16 + ln];
    float bh = bhh[128 + jj * 16 + ln];
    #pragma unroll
    for (int e = 0; e < 8; ++e) { accI[jj][e] = bi; accH[jj][e] = bh; }
  }

  #pragma unroll
  for (int kt = 0; kt < 8; ++kt) {          // K = 256, 32 per WMMA
    const int K0 = kt * 32 + kb;
    v8bf a0, a1;
    if (kt < 6) {                           // x-neighborhood region (K<192)
      int dy0 = K0 >> 6,        c0 = K0 & 63;
      int dy1 = (K0 + 16) >> 6, c1 = (K0 + 16) & 63;
      a0 = *(const v8bf*)(sPrev + (rowA + dy0) * LDS_ROW + c0);
      a1 = *(const v8bf*)(sPrev + (rowA + dy1) * LDS_ROW + c1);
    } else {                                // h region (K>=192)
      a0 = *(const v8bf*)(sH + rowA * LDS_ROW + (K0 - 192));
      a1 = *(const v8bf*)(sH + rowA * LDS_ROW + (K0 - 192 + 16));
    }
    v16bf A = __builtin_shufflevector(a0, a1, 0,1,2,3,4,5,6,7,8,9,10,11,12,13,14,15);
    #pragma unroll
    for (int n = 0; n < 12; ++n) {
      const unsigned short* wrow = Wc + (size_t)(n * 16 + ln) * KTOT + kt * 32 + kb;
      v8bf b0 = *(const v8bf*)(wrow);
      v8bf b1 = *(const v8bf*)(wrow + 16);
      v16bf Bm = __builtin_shufflevector(b0, b1, 0,1,2,3,4,5,6,7,8,9,10,11,12,13,14,15);
      if (n < 8)
        accRZ[n] = __builtin_amdgcn_wmma_f32_16x16x32_bf16(
            false, A, false, Bm, (short)0, accRZ[n], false, false);
      else if (kt < 6)
        accI[n - 8] = __builtin_amdgcn_wmma_f32_16x16x32_bf16(
            false, A, false, Bm, (short)0, accI[n - 8], false, false);
      else
        accH[n - 8] = __builtin_amdgcn_wmma_f32_16x16x32_bf16(
            false, A, false, Bm, (short)0, accH[n - 8], false, false);
    }
  }

  // GRU elementwise: r=sig(g0..63) z=sig(g64..127) n=tanh(i_n + r*h_n)
  #pragma unroll
  for (int j = 0; j < 4; ++j) {
    v8f r, z, nn;
    #pragma unroll
    for (int e = 0; e < 8; ++e) {
      r[e] = 1.f / (1.f + __expf(-accRZ[j][e]));
      z[e] = 1.f / (1.f + __expf(-accRZ[4 + j][e]));
      float pre = accI[j][e] + r[e] * accH[j][e];
      nn[e] = 1.f - 2.f / (__expf(2.f * pre) + 1.f);   // tanh, overflow-safe
    }
    const int c = j * 16 + ln;
    #pragma unroll
    for (int e = 0; e < 8; ++e) {
      int rloc = mloc + e + 8 * half;                  // C-layout row (ISA table)
      float h  = bf2f(sH[rloc * LDS_ROW + c]);
      float ov = (1.f - z[e]) * nn[e] + z[e] * h;
      int m = rowBase + rloc;
      int b = m >> 9, hh = m & (Hh - 1);
      out[(((size_t)(b * Cc + c) * Hh) + hh) * Ww + d] = ov;
      sH[rloc * LDS_ROW + c] = f2bf(ov);               // updated state (own rows)
    }
  }
  __syncthreads();
  // Coalesced write-back of updated column state into xT[d]
  for (int i = t; i < ROWS_PER_WG * 8; i += 256) {
    int r = i >> 3, kc = (i & 7) * 8;
    *(uint4*)(curCol + (size_t)(rowBase + r) * Cc + kc) =
        *(const uint4*)(sH + r * LDS_ROW + kc);
  }
}

// ---------------------------------------------------------------------------
extern "C" void kernel_launch(void* const* d_in, const int* in_sizes, int n_in,
                              void* d_out, int out_size, void* d_ws, size_t ws_size,
                              hipStream_t stream) {
  (void)in_sizes; (void)n_in; (void)out_size; (void)ws_size;
  const float* feat = (const float*)d_in[0];   // [4,64,512,512]
  const float* wih  = (const float*)d_in[1];   // [192,192]
  const float* whh  = (const float*)d_in[2];   // [192,64]
  const float* bih  = (const float*)d_in[3];   // [192]
  const float* bhh  = (const float*)d_in[4];   // [192]
  float* out = (float*)d_out;

  // Workspace: Wcomb bf16 (96 KB, padded to 128 KB) + xT bf16 (128 MB)
  unsigned short* Wc = (unsigned short*)d_ws;
  unsigned short* xT = (unsigned short*)((char*)d_ws + (128u << 10));

  gru_prep_weights<<<(GATES * KTOT + 255) / 256, 256, 0, stream>>>(wih, whh, Wc);

  dim3 tg(Ww / 32, MT);
  gru_transpose<<<tg, 256, 0, stream>>>(feat, xT, out);

  for (int d = 1; d < Ww; ++d)
    gru_scan_step<<<NUM_WG, 256, 0, stream>>>(Wc, xT, bih, bhh, out, d);
}
